// HyperbolicTemporalTransformer_19516331393625
// MI455X (gfx1250) — matrix operand order repaired and checked by reading, not compiled, not run on hardware
//
#include <hip/hip_runtime.h>
#include <hip/hip_bf16.h>

// ---------------- model constants ----------------
#define NN      40000
#define NE      640000
#define DIM     128      // IN_DIM == HID == OUT_DIM
#define HEADS   4
#define HD      32
#define NLAYERS 2
#define EDIM    16

typedef __attribute__((ext_vector_type(16))) _Float16 v16h;
typedef __attribute__((ext_vector_type(8)))  _Float16 v8h;
typedef __attribute__((ext_vector_type(8)))  float    v8f;
typedef __attribute__((ext_vector_type(4)))  unsigned u32x4;
typedef __attribute__((ext_vector_type(8)))  int      i32x8;
typedef __attribute__((ext_vector_type(4)))  int      i32x4;

#if defined(__gfx1250__) && __has_builtin(__builtin_amdgcn_tensor_load_to_lds) && \
    __has_builtin(__builtin_amdgcn_s_wait_tensorcnt)
#define USE_TDM 1
#else
#define USE_TDM 0
#endif

// ordered-uint mapping so unsigned atomicMax == float max
__device__ __forceinline__ unsigned f2o(float f) {
    unsigned u = __float_as_uint(f);
    return (u & 0x80000000u) ? ~u : (u | 0x80000000u);
}
__device__ __forceinline__ float o2f(unsigned o) {
    unsigned u = (o & 0x80000000u) ? (o & 0x7fffffffu) : ~o;
    return __uint_as_float(u);
}

// ---------------- weight prep: strip time column, f16-ize ----------------
__global__ void prep_w_kernel(const float* __restrict__ src,
                              _Float16* __restrict__ dstW,
                              float* __restrict__ dstw0) {
    int idx = blockIdx.x * blockDim.x + threadIdx.x;
    if (idx >= DIM * DIM) return;
    int o = idx >> 7, k = idx & 127;
    dstW[idx] = (_Float16)src[o * (DIM + 1) + 1 + k];
    if (k == 0) dstw0[o] = src[o * (DIM + 1)];
}

__global__ void zero_kernel(float* __restrict__ p, int count) {
    int i = blockIdx.x * blockDim.x + threadIdx.x;
    if (i < count) p[i] = 0.0f;
}

// ---------------- WMMA GEMM: Y[m,o] = t[m]*w0[o] + sum_k A[m,k]*W[o,k] + b[o] ----------------
// Block = 256 threads = 8 waves; each wave owns 32 rows (2 M-tiles) x 128 cols.
// W (128x128 f16, 32KB) staged in LDS via Tensor Data Mover (wave 0), fallback uint4 copy.
__global__ __launch_bounds__(256) void gemm_kernel(
    const _Float16* __restrict__ A, const float* __restrict__ tvec,
    const _Float16* __restrict__ W, const float* __restrict__ w0,
    const float* __restrict__ bias, float* __restrict__ Y, int nrows)
{
    __shared__ _Float16 sW[DIM * DIM];
    int tid = threadIdx.x;

#if USE_TDM
    if (tid < 32) {
        // Tensor DMA descriptor (D#): 2D tensor, 2-byte elements, 128x128 tile == tensor
        unsigned lds = (unsigned)(size_t)(&sW[0]);
        unsigned long long ga = (unsigned long long)(size_t)W;
        u32x4 g0 = { 1u,                                  // count=1, user mode
                     lds,                                 // lds_addr
                     (unsigned)ga,                        // global_addr[31:0]
                     (unsigned)(ga >> 32) | (2u << 30) }; // global_addr[56:32] | type=2
        i32x8 g1;
        g1[0] = (int)(1u << 16);          // wg_mask=0 | data_size=1 (2 bytes)
        g1[1] = (int)(128u << 16);        // barrier_addr=0 | tensor_dim0[15:0]=128
        g1[2] = (int)(128u << 16);        // tensor_dim0[31:16]=0 | tensor_dim1[15:0]=128
        g1[3] = (int)(128u << 16);        // tensor_dim1[31:16]=0 | tile_dim0=128
        g1[4] = (int)128;                 // tile_dim1=128 | tile_dim2=0
        g1[5] = (int)128;                 // tensor_dim0_stride[31:0]=128
        g1[6] = 0;                        // stride hi | dim1_stride lo
        g1[7] = 0;
        i32x4 gz4 = {0, 0, 0, 0};
        i32x8 gz8 = {0, 0, 0, 0, 0, 0, 0, 0};
        __builtin_amdgcn_tensor_load_to_lds(g0, g1, gz4, gz4, gz8, 0);
        __builtin_amdgcn_s_wait_tensorcnt((short)0);
    }
    __syncthreads();
#else
    {
        const uint4* wsrc = (const uint4*)W;
        uint4*       wdst = (uint4*)sW;
        #pragma unroll
        for (int i = 0; i < 8; ++i) wdst[tid + i * 256] = wsrc[tid + i * 256];
    }
    __syncthreads();
#endif

    int wave = tid >> 5;
    int lane = tid & 31;
    int hi   = lane >> 4;   // 0 | 1
    int l16  = lane & 15;
    int rowbase = blockIdx.x * 256 + wave * 32;

    // C layout: vgpr r -> M = r + 8*hi, N = l16 (+16*nt)
    v8f acc0[8], acc1[8];
    #pragma unroll
    for (int nt = 0; nt < 8; ++nt) {
        int n = nt * 16 + l16;
        float w0n = w0[n];
        float bn  = bias[n];
        #pragma unroll
        for (int r = 0; r < 8; ++r) {
            int m0 = rowbase + hi * 8 + r;
            int m1 = m0 + 16;
            acc0[nt][r] = tvec[m0 < nrows ? m0 : 0] * w0n + bn;
            acc1[nt][r] = tvec[m1 < nrows ? m1 : 0] * w0n + bn;
        }
    }

    int arow0 = rowbase + l16;       if (arow0 >= nrows) arow0 = nrows - 1;
    int arow1 = rowbase + 16 + l16;  if (arow1 >= nrows) arow1 = nrows - 1;
    const _Float16* A0 = A + (size_t)arow0 * DIM;
    const _Float16* A1 = A + (size_t)arow1 * DIM;
    __builtin_prefetch(A0, 0, 3);    // global_prefetch_b8
    __builtin_prefetch(A1, 0, 3);
    int aoff = hi * 8;               // A layout: lanes<16 K 0..7/16..23, lanes>=16 K 8..15/24..31

    #pragma unroll
    for (int kt = 0; kt < 4; ++kt) {
        int kb = kt * 32;
        v8h a0lo = *(const v8h*)(A0 + kb + aoff);
        v8h a0hi = *(const v8h*)(A0 + kb + 16 + aoff);
        v8h a1lo = *(const v8h*)(A1 + kb + aoff);
        v8h a1hi = *(const v8h*)(A1 + kb + 16 + aoff);
        v16h a0, a1;
        #pragma unroll
        for (int i = 0; i < 8; ++i) {
            a0[i] = a0lo[i]; a0[i + 8] = a0hi[i];
            a1[i] = a1lo[i]; a1[i + 8] = a1hi[i];
        }

        int koff = kb + hi * 8;      // B layout: lanes<16 K 0..7/16..23, lanes>=16 K 8..15/24..31, col = l16
        v16h bcur;
        {
            v8h blo = *(const v8h*)(sW + (size_t)l16 * DIM + koff);
            v8h bhi = *(const v8h*)(sW + (size_t)l16 * DIM + koff + 16);
            #pragma unroll
            for (int i = 0; i < 8; ++i) { bcur[i] = blo[i]; bcur[i + 8] = bhi[i]; }
        }
        #pragma unroll
        for (int nt = 0; nt < 8; ++nt) {
            v16h bnext = bcur;
            if (nt < 7) {
                v8h blo = *(const v8h*)(sW + (size_t)((nt + 1) * 16 + l16) * DIM + koff);
                v8h bhi = *(const v8h*)(sW + (size_t)((nt + 1) * 16 + l16) * DIM + koff + 16);
                #pragma unroll
                for (int i = 0; i < 8; ++i) { bnext[i] = blo[i]; bnext[i + 8] = bhi[i]; }
            }
            acc0[nt] = __builtin_amdgcn_wmma_f32_16x16x32_f16(
                false, a0, false, bcur, (short)0, acc0[nt], false, false);
            acc1[nt] = __builtin_amdgcn_wmma_f32_16x16x32_f16(
                false, a1, false, bcur, (short)0, acc1[nt], false, false);
            bcur = bnext;
        }
    }

    #pragma unroll
    for (int r = 0; r < 8; ++r) {
        int m0 = rowbase + hi * 8 + r;
        int m1 = m0 + 16;
        if (m0 < nrows) {
            float* yrow = Y + (size_t)m0 * DIM;
            #pragma unroll
            for (int nt = 0; nt < 8; ++nt) yrow[nt * 16 + l16] = acc0[nt][r];
        }
        if (m1 < nrows) {
            float* yrow = Y + (size_t)m1 * DIM;
            #pragma unroll
            for (int nt = 0; nt < 8; ++nt) yrow[nt * 16 + l16] = acc1[nt][r];
        }
    }
}

// ---------------- node kernels ----------------
__global__ void node_pre_kernel(const float* __restrict__ x,
                                float* __restrict__ tA, _Float16* __restrict__ Af16) {
    int n = blockIdx.x * blockDim.x + threadIdx.x;
    if (n >= NN) return;
    const float* xr = x + (size_t)n * DIM;
    float s = 0.f;
    for (int k = 0; k < DIM; ++k) s += xr[k] * xr[k];
    float nn = sqrtf(fmaxf(s, 1e-12f));
    tA[n] = coshf(nn);
    float sc = sinhf(nn) / nn;
    _Float16* ar = Af16 + (size_t)n * DIM;
    for (int k = 0; k < DIM; ++k) ar[k] = (_Float16)(xr[k] * sc);
}

__global__ void node_post_in_kernel(const float* __restrict__ Y,
                                    const float* __restrict__ g, const float* __restrict__ b,
                                    float* __restrict__ H, _Float16* __restrict__ Hf16,
                                    float* __restrict__ tH) {
    int n = blockIdx.x * blockDim.x + threadIdx.x;
    if (n >= NN) return;
    const float* yr = Y + (size_t)n * DIM;
    float su = 0.f, sq = 0.f;
    for (int k = 0; k < DIM; ++k) { float v = yr[k]; su += v; sq += v * v; }
    float mu  = su / DIM;
    float var = sq / DIM - mu * mu;
    float inv = rsqrtf(var + 1e-5f);
    float s2 = 0.f;
    float* hr = H + (size_t)n * DIM;
    _Float16* hf = Hf16 + (size_t)n * DIM;
    for (int k = 0; k < DIM; ++k) {
        float v = (yr[k] - mu) * inv * g[k] + b[k];
        v = fmaxf(v, 0.f);
        hr[k] = v; hf[k] = (_Float16)v; s2 += v * v;
    }
    tH[n] = sqrtf(1.f + s2);
}

__global__ void qkv_time_kernel(const float* __restrict__ Q, const float* __restrict__ K,
                                const float* __restrict__ V,
                                float* __restrict__ qt, float* __restrict__ kt,
                                float* __restrict__ vt) {
    int idx = blockIdx.x * blockDim.x + threadIdx.x;
    if (idx >= NN * HEADS) return;
    int n = idx >> 2, h = idx & 3;
    size_t base = (size_t)n * DIM + h * HD;
    float sq = 0.f, sk = 0.f, sv = 0.f;
    for (int j = 0; j < HD; ++j) {
        float q = Q[base + j], k = K[base + j], v = V[base + j];
        sq += q * q; sk += k * k; sv += v * v;
    }
    qt[idx] = sqrtf(1.f + sq);
    kt[idx] = sqrtf(1.f + sk);
    vt[idx] = sqrtf(1.f + sv);
}

// ---------------- edge kernels (segment softmax by atomics) ----------------
__global__ void edge_scores_kernel(const int* __restrict__ ei, const float* __restrict__ ef,
                                   const float* __restrict__ Q, const float* __restrict__ K,
                                   const float* __restrict__ qt, const float* __restrict__ kt,
                                   const float* __restrict__ ebw,
                                   float* __restrict__ scores, unsigned* __restrict__ mbuf) {
    int idx = blockIdx.x * blockDim.x + threadIdx.x;
    if (idx >= NE * HEADS) return;
    int e = idx >> 2, h = idx & 3;
    int src = ei[e], dst = ei[NE + e];
    const float* qr = Q + (size_t)dst * DIM + h * HD;
    const float* kr = K + (size_t)src * DIM + h * HD;
    float d = 0.f;
    for (int j = 0; j < HD; ++j) d += qr[j] * kr[j];
    float c = d - qt[dst * HEADS + h] * kt[src * HEADS + h];
    float s = (2.f + 2.f * c) * 0.17677669529663687f;   // /sqrt(32)
    const float* ew = ebw + h * EDIM;
    const float* er = ef + (size_t)e * EDIM;
    float eb = 0.f;
    for (int j = 0; j < EDIM; ++j) eb += er[j] * ew[j];
    s += eb;
    scores[idx] = s;
    atomicMax(&mbuf[dst * HEADS + h], f2o(s));
}

__global__ void edge_exp_kernel(const int* __restrict__ ei,
                                const unsigned* __restrict__ mbuf,
                                float* __restrict__ scores, float* __restrict__ denom) {
    int idx = blockIdx.x * blockDim.x + threadIdx.x;
    if (idx >= NE * HEADS) return;
    int e = idx >> 2, h = idx & 3;
    int dst = ei[NE + e];
    float m = o2f(mbuf[dst * HEADS + h]);
    float e2 = expf(scores[idx] - m);
    scores[idx] = e2;
    atomicAdd(&denom[dst * HEADS + h], e2);
}

__global__ void edge_accum_kernel(const int* __restrict__ ei,
                                  const float* __restrict__ scores,
                                  const float* __restrict__ denom,
                                  const float* __restrict__ V, const float* __restrict__ vt,
                                  float* __restrict__ ave) {
    int idx = blockIdx.x * blockDim.x + threadIdx.x;
    if (idx >= NE * HEADS) return;
    int e = idx >> 2, h = idx & 3;
    int src = ei[e], dst = ei[NE + e];
    float a = scores[idx] / (denom[dst * HEADS + h] + 1e-16f);
    float* av = ave + (size_t)dst * (HEADS * (HD + 1)) + h * (HD + 1);
    atomicAdd(av, a * vt[src * HEADS + h]);
    const float* vr = V + (size_t)src * DIM + h * HD;
    for (int j = 0; j < HD; ++j) atomicAdd(av + 1 + j, a * vr[j]);
}

__global__ void node_attn_out_kernel(const float* __restrict__ ave,
                                     _Float16* __restrict__ Of16, float* __restrict__ tO) {
    int n = blockIdx.x * blockDim.x + threadIdx.x;
    if (n >= NN) return;
    _Float16* orow = Of16 + (size_t)n * DIM;
    float s2 = 0.f;
    for (int h = 0; h < HEADS; ++h) {
        const float* av = ave + (size_t)n * (HEADS * (HD + 1)) + h * (HD + 1);
        float at = av[0], ss = 0.f;
        for (int j = 0; j < HD; ++j) { float v = av[1 + j]; ss += v * v; }
        float den = sqrtf(fmaxf(fabsf(at * at - ss), 1e-8f));
        float invd = 1.f / den;
        for (int j = 0; j < HD; ++j) {
            float o = av[1 + j] * invd;
            s2 += o * o;
            orow[h * HD + j] = (_Float16)o;
        }
    }
    tO[n] = sqrtf(1.f + s2);
}

__global__ void node_mid_norm_kernel(const float* __restrict__ Z,
                                     const float* __restrict__ g, const float* __restrict__ b,
                                     float* __restrict__ H, _Float16* __restrict__ Hf16,
                                     float* __restrict__ tH) {
    int n = blockIdx.x * blockDim.x + threadIdx.x;
    if (n >= NN) return;
    const float* zr = Z + (size_t)n * DIM;
    float* hr = H + (size_t)n * DIM;
    float sz = 0.f, sa = 0.f, ssum = 0.f;
    for (int k = 0; k < DIM; ++k) {
        float z = zr[k], hh = hr[k];
        sz += z * z;
        float av = 0.5f * (z + hh);
        sa += av * av; ssum += av;
    }
    float a_t = sqrtf(1.f + sz);
    float b_t = tH[n];
    float ave_t = 0.5f * (a_t + b_t);
    float den = sqrtf(fmaxf(fabsf(ave_t * ave_t - sa), 1e-8f));
    float invd = 1.f / den;
    float mean = ssum * invd / DIM;
    float var  = sa * invd * invd / DIM - mean * mean;
    float invs = rsqrtf(var + 1e-5f);
    float s2 = 0.f;
    _Float16* hf = Hf16 + (size_t)n * DIM;
    for (int k = 0; k < DIM; ++k) {
        float av = 0.5f * (zr[k] + hr[k]);
        float v = (av * invd - mean) * invs * g[k] + b[k];
        hr[k] = v; hf[k] = (_Float16)v; s2 += v * v;
    }
    tH[n] = sqrtf(1.f + s2);
}

__global__ void node_final_kernel(const float* __restrict__ Y, float* __restrict__ out) {
    int n = blockIdx.x * blockDim.x + threadIdx.x;
    if (n >= NN) return;
    const float* yr = Y + (size_t)n * DIM;
    float s = 0.f;
    for (int k = 0; k < DIM; ++k) s += yr[k] * yr[k];
    float ns = sqrtf(fmaxf(s, 1e-12f));
    float t  = sqrtf(1.f + s);
    float alpha = fmaxf(t, 1.f + 1e-7f);
    float sc = acoshf(alpha) / ns;
    float* orow = out + (size_t)n * DIM;
    for (int k = 0; k < DIM; ++k) orow[k] = yr[k] * sc;
}

// ---------------- host side ----------------
extern "C" void kernel_launch(void* const* d_in, const int* in_sizes, int n_in,
                              void* d_out, int out_size, void* d_ws, size_t ws_size,
                              hipStream_t stream) {
    const float* x      = (const float*)d_in[0];
    const int*   ei     = (const int*)d_in[1];
    const float* ef     = (const float*)d_in[2];
    const float* in_w   = (const float*)d_in[3];
    const float* in_b   = (const float*)d_in[4];
    const float* in_g   = (const float*)d_in[5];
    const float* in_bb  = (const float*)d_in[6];
    const float* q_w    = (const float*)d_in[7];
    const float* q_b    = (const float*)d_in[8];
    const float* k_w    = (const float*)d_in[9];
    const float* k_b    = (const float*)d_in[10];
    const float* v_w    = (const float*)d_in[11];
    const float* v_b    = (const float*)d_in[12];
    const float* out_w  = (const float*)d_in[13];
    const float* out_b  = (const float*)d_in[14];
    const float* eb_w   = (const float*)d_in[15];
    const float* norm_g = (const float*)d_in[16];
    const float* norm_b = (const float*)d_in[17];
    const float* outp_w = (const float*)d_in[18];
    const float* outp_b = (const float*)d_in[19];
    float* outp = (float*)d_out;

    char* wsp = (char*)d_ws;
    auto carve = [&](size_t bytes) -> void* {
        void* p = (void*)wsp;
        wsp += (bytes + 255) & ~(size_t)255;
        return p;
    };
    _Float16* Wf16 = (_Float16*)carve((size_t)10 * DIM * DIM * sizeof(_Float16));
    float*    w0m  = (float*)carve((size_t)10 * DIM * sizeof(float));
    float*    tH   = (float*)carve((size_t)NN * sizeof(float));
    _Float16* Hf16 = (_Float16*)carve((size_t)NN * DIM * sizeof(_Float16));
    float*    H    = (float*)carve((size_t)NN * DIM * sizeof(float));
    float*    tA   = (float*)carve((size_t)NN * sizeof(float));
    _Float16* Af16 = (_Float16*)carve((size_t)NN * DIM * sizeof(_Float16));
    float*    Qb   = (float*)carve((size_t)NN * DIM * sizeof(float));
    float*    Kb   = (float*)carve((size_t)NN * DIM * sizeof(float));
    float*    Vb   = (float*)carve((size_t)NN * DIM * sizeof(float));
    float*    qt   = (float*)carve((size_t)NN * HEADS * sizeof(float));
    float*    kt   = (float*)carve((size_t)NN * HEADS * sizeof(float));
    float*    vt   = (float*)carve((size_t)NN * HEADS * sizeof(float));
    float*    scor = (float*)carve((size_t)NE * HEADS * sizeof(float));
    float*    mreg = (float*)carve((size_t)(NN * HEADS + NN * HEADS + NN * HEADS * (HD + 1)) * sizeof(float));
    unsigned* mbuf = (unsigned*)mreg;
    float*    denom = mreg + NN * HEADS;
    float*    ave   = mreg + 2 * NN * HEADS;
    float*    Y    = (float*)carve((size_t)NN * DIM * sizeof(float));
    (void)ws_size; (void)in_sizes; (void)n_in; (void)out_size;

    const int TB = 256;
    const int nodeB = (NN + TB - 1) / TB;
    const int nhB   = (NN * HEADS + TB - 1) / TB;
    const int ehB   = (NE * HEADS + TB - 1) / TB;
    const int gemmB = (NN + 255) / 256;     // 32 rows per wave, 8 waves per block
    const int prepB = (DIM * DIM + TB - 1) / TB;
    const int zeroCount = NN * (2 * HEADS + HEADS * (HD + 1));
    const int zeroB = (zeroCount + TB - 1) / TB;

    const float* wsrc[10];
    wsrc[0] = in_w;
    for (int l = 0; l < NLAYERS; ++l) {
        wsrc[1 + 4 * l] = q_w + (size_t)l * DIM * (DIM + 1);
        wsrc[2 + 4 * l] = k_w + (size_t)l * DIM * (DIM + 1);
        wsrc[3 + 4 * l] = v_w + (size_t)l * DIM * (DIM + 1);
        wsrc[4 + 4 * l] = out_w + (size_t)l * DIM * (DIM + 1);
    }
    wsrc[9] = outp_w;
    for (int mi = 0; mi < 10; ++mi)
        prep_w_kernel<<<prepB, TB, 0, stream>>>(wsrc[mi], Wf16 + (size_t)mi * DIM * DIM,
                                                w0m + (size_t)mi * DIM);

    node_pre_kernel<<<nodeB, TB, 0, stream>>>(x, tA, Af16);
    gemm_kernel<<<gemmB, TB, 0, stream>>>(Af16, tA, Wf16, w0m, in_b, Y, NN);
    node_post_in_kernel<<<nodeB, TB, 0, stream>>>(Y, in_g, in_bb, H, Hf16, tH);

    for (int l = 0; l < NLAYERS; ++l) {
        const _Float16* Wq = Wf16 + (size_t)(1 + 4 * l) * DIM * DIM;
        const _Float16* Wk = Wf16 + (size_t)(2 + 4 * l) * DIM * DIM;
        const _Float16* Wv = Wf16 + (size_t)(3 + 4 * l) * DIM * DIM;
        const _Float16* Wo = Wf16 + (size_t)(4 + 4 * l) * DIM * DIM;
        gemm_kernel<<<gemmB, TB, 0, stream>>>(Hf16, tH, Wq, w0m + (1 + 4 * l) * DIM,
                                              q_b + l * DIM, Qb, NN);
        gemm_kernel<<<gemmB, TB, 0, stream>>>(Hf16, tH, Wk, w0m + (2 + 4 * l) * DIM,
                                              k_b + l * DIM, Kb, NN);
        gemm_kernel<<<gemmB, TB, 0, stream>>>(Hf16, tH, Wv, w0m + (3 + 4 * l) * DIM,
                                              v_b + l * DIM, Vb, NN);
        qkv_time_kernel<<<nhB, TB, 0, stream>>>(Qb, Kb, Vb, qt, kt, vt);

        zero_kernel<<<zeroB, TB, 0, stream>>>(mreg, zeroCount);
        edge_scores_kernel<<<ehB, TB, 0, stream>>>(ei, ef, Qb, Kb, qt, kt,
                                                   eb_w + l * HEADS * EDIM, scor, mbuf);
        edge_exp_kernel<<<ehB, TB, 0, stream>>>(ei, mbuf, scor, denom);
        edge_accum_kernel<<<ehB, TB, 0, stream>>>(ei, scor, denom, Vb, vt, ave);

        node_attn_out_kernel<<<nodeB, TB, 0, stream>>>(ave, Af16, tA);
        gemm_kernel<<<gemmB, TB, 0, stream>>>(Af16, tA, Wo, w0m + (4 + 4 * l) * DIM,
                                              out_b + l * DIM, Y, NN);
        node_mid_norm_kernel<<<nodeB, TB, 0, stream>>>(Y, norm_g + l * DIM, norm_b + l * DIM,
                                                       H, Hf16, tH);
    }

    gemm_kernel<<<gemmB, TB, 0, stream>>>(Hf16, tH, Wf16 + (size_t)9 * DIM * DIM,
                                          w0m + 9 * DIM, outp_b, Y, NN);
    node_final_kernel<<<nodeB, TB, 0, stream>>>(Y, outp);
}
